// Gemma4MoE_38019050504858
// MI455X (gfx1250) — matrix operand order repaired and checked by reading, not compiled, not run on hardware
//
#include <hip/hip_runtime.h>
#include <hip/hip_bf16.h>
#include <math.h>

// ---------------------------------------------------------------------------
// Gemma4 MoE for MI455X (gfx1250, wave32, WMMA).
//   T=8192 tokens, H=2048 hidden, E=8 experts, I=2048 intermediate, top-2.
// Dense-with-gating (gate==0 kills non-selected experts, matching reference).
// ~1.65 TFLOP bf16 vs ~1.5GB HBM traffic @ 23.3TB/s => compute-bound, so the
// design centers on v_wmma_f32_16x16x32_bf16 fed from LDS, with the stage-B
// A-tile staged via global_load_async_to_lds_b128 (ASYNCcnt path).
// Per-expert hbuf (33.5MB bf16) is sized to sit in the 192MB L2.
// ---------------------------------------------------------------------------

#define T_TOK 8192
#define HDIM  2048
#define NEXP  8
#define IDIM  2048

typedef __attribute__((ext_vector_type(16))) __bf16 v16bf;
typedef __attribute__((ext_vector_type(8)))  __bf16 v8bf;
typedef __attribute__((ext_vector_type(4)))  __bf16 v4bf;
typedef __attribute__((ext_vector_type(8)))  float  v8f;
typedef __attribute__((ext_vector_type(4)))  float  f4;

union FragU {            // 16x32 bf16 fragment: two 16B halves per lane
  v16bf v;
  v8bf  h[2];
};

// ------------------------------ router -------------------------------------
__device__ __forceinline__ float wave_sum(float v) {
#pragma unroll
  for (int off = 16; off > 0; off >>= 1) v += __shfl_xor(v, off, 32);
  return v;
}

__global__ void router_kernel(const float* __restrict__ ri,
                              const float* __restrict__ scale,
                              const float* __restrict__ pw,    // [E,H]
                              const float* __restrict__ pes,   // [E]
                              float* __restrict__ gate) {      // [T,E] dense
  const int wave = threadIdx.x >> 5;
  const int lane = threadIdx.x & 31;
  const int t = blockIdx.x * (blockDim.x >> 5) + wave;
  if (t >= T_TOK) return;
  const float* row = ri + (size_t)t * HDIM;

  // RMS (no weight)
  float ss = 0.f;
  for (int h = lane; h < HDIM; h += 32) { float x = row[h]; ss += x * x; }
  ss = wave_sum(ss);
  const float rms    = rsqrtf(ss * (1.0f / HDIM) + 1e-6f);
  const float cscale = rms * 0.022097086912079608f;  // * H^-0.5

  // 8 router dots
  float acc[NEXP];
#pragma unroll
  for (int e = 0; e < NEXP; ++e) acc[e] = 0.f;
  for (int h = lane; h < HDIM; h += 32) {
    const float x = row[h] * cscale * scale[h];
#pragma unroll
    for (int e = 0; e < NEXP; ++e) acc[e] += x * pw[e * HDIM + h];
  }
#pragma unroll
  for (int e = 0; e < NEXP; ++e) acc[e] = wave_sum(acc[e]);

  // softmax over 8 (all lanes redundantly)
  float mx = acc[0];
#pragma unroll
  for (int e = 1; e < NEXP; ++e) mx = fmaxf(mx, acc[e]);
  float p[NEXP]; float sum = 0.f;
#pragma unroll
  for (int e = 0; e < NEXP; ++e) { p[e] = __expf(acc[e] - mx); sum += p[e]; }
  const float inv = 1.f / sum;
#pragma unroll
  for (int e = 0; e < NEXP; ++e) p[e] *= inv;

  // top-2, renorm, per-expert scale
  int i1 = 0; float p1 = p[0];
#pragma unroll
  for (int e = 1; e < NEXP; ++e) if (p[e] > p1) { p1 = p[e]; i1 = e; }
  int i2 = (i1 == 0) ? 1 : 0; float p2 = p[i2];
#pragma unroll
  for (int e = 0; e < NEXP; ++e)
    if (e != i1 && p[e] > p2) { p2 = p[e]; i2 = e; }
  const float rs = 1.f / (p1 + p2 + 1e-20f);
  const float g1 = p1 * rs * pes[i1];
  const float g2 = p2 * rs * pes[i2];
  if (lane < NEXP)
    gate[(size_t)t * NEXP + lane] = (lane == i1) ? g1 : (lane == i2) ? g2 : 0.f;
}

// ------------------------------ GEMM tiles ---------------------------------
#define BM 128
#define BN 128
#define BK 64
#define LSTR 72   // LDS row stride (elems): rows stay 16B aligned, skewed

// A fragment (16x32 bf16) from LDS [m][k] tile.
// ISA A layout: lane<16 -> K {kk..kk+7, kk+16..kk+23}; lane>=16 -> +8.
__device__ __forceinline__ FragU load_frag_a(const __bf16* base, int row, int kk,
                                             int lane) {
  FragU f;
  const __bf16* p = base + row * LSTR + kk + ((lane >> 4) << 3);
  f.h[0] = *(const v8bf*)p;
  f.h[1] = *(const v8bf*)(p + 16);
  return f;
}

// B fragment (32x16 bf16) from LDS [n][k] (transposed) tile.
// ISA B layout: lane<16 -> K {kk..kk+7, kk+16..kk+23} for col n=lane; lane>=16 -> +8.
__device__ __forceinline__ FragU load_frag_b(const __bf16* base, int col, int kk,
                                             int lane) {
  FragU f;
  const __bf16* p = base + col * LSTR + kk + ((lane >> 4) << 3);
  f.h[0] = *(const v8bf*)p;
  f.h[1] = *(const v8bf*)(p + 16);
  return f;
}

// ---------------- stage A: h = gelu(X@W1) * (X@W3), bf16 out ---------------
__global__ __launch_bounds__(256)
void geglu_up_kernel(const float* __restrict__ X,    // [T,H]
                     const float* __restrict__ W1,   // [H,I]
                     const float* __restrict__ W3,   // [H,I]
                     __bf16* __restrict__ hbuf) {    // [T,I]
  __shared__ alignas(16) __bf16 As[BM * LSTR];       // [m][k]
  __shared__ alignas(16) __bf16 Bg[BN * LSTR];       // [n][k] (transposed)
  __shared__ alignas(16) __bf16 Bu[BN * LSTR];

  const int tid  = threadIdx.x;
  const int lane = tid & 31;
  const int wave = tid >> 5;
  const int wm = (wave >> 1) * 32;
  const int wn = (wave & 1) * 64;
  const int m0 = blockIdx.y * BM;
  const int n0 = blockIdx.x * BN;

  const v8f vzero = {0.f,0.f,0.f,0.f,0.f,0.f,0.f,0.f};
  v8f accG[2][4], accU[2][4];
#pragma unroll
  for (int i = 0; i < 2; ++i)
#pragma unroll
    for (int j = 0; j < 4; ++j) { accG[i][j] = vzero; accU[i][j] = vzero; }

  for (int k0 = 0; k0 < HDIM; k0 += BK) {
    // stage A tile 128x64 (f32 -> bf16), b64 LDS stores
#pragma unroll
    for (int i = 0; i < 8; ++i) {
      int lin = i * 256 + tid;            // 2048 float4 slots
      int r = lin >> 4;                   // BK/4 = 16 slots per row
      int c = (lin & 15) << 2;
      f4 x = *(const f4*)&X[(size_t)(m0 + r) * HDIM + k0 + c];
      v4bf b; b[0] = (__bf16)x[0]; b[1] = (__bf16)x[1];
              b[2] = (__bf16)x[2]; b[3] = (__bf16)x[3];
      *(v4bf*)&As[r * LSTR + c] = b;
    }
    // stage B tiles 64x128, stored transposed [n][k]
#pragma unroll
    for (int i = 0; i < 8; ++i) {
      int lin = i * 256 + tid;            // 2048 float4 slots
      int k = lin >> 5;                   // BN/4 = 32 slots per k-row
      int c = (lin & 31) << 2;
      size_t g = (size_t)(k0 + k) * IDIM + n0 + c;
      f4 a = *(const f4*)&W1[g];
      f4 u = *(const f4*)&W3[g];
#pragma unroll
      for (int j = 0; j < 4; ++j) {
        Bg[(c + j) * LSTR + k] = (__bf16)a[j];
        Bu[(c + j) * LSTR + k] = (__bf16)u[j];
      }
    }
    if (k0 + BK < HDIM) {   // pull next K-tile toward L2 (global_prefetch_b8)
      __builtin_prefetch(&X[(size_t)(m0 + (tid >> 1)) * HDIM + k0 + BK], 0, 1);
      __builtin_prefetch(&W1[(size_t)(k0 + BK + (tid >> 2)) * IDIM + n0], 0, 1);
      __builtin_prefetch(&W3[(size_t)(k0 + BK + (tid >> 2)) * IDIM + n0], 0, 1);
    }
    __syncthreads();

#pragma unroll
    for (int kk = 0; kk < BK; kk += 32) {
      const int lm = lane & 15;
      FragU a0 = load_frag_a(As, wm + lm,      kk, lane);
      FragU a1 = load_frag_a(As, wm + 16 + lm, kk, lane);
#pragma unroll
      for (int nt = 0; nt < 4; ++nt) {
        FragU bg = load_frag_b(Bg, wn + nt * 16 + lm, kk, lane);
        FragU bu = load_frag_b(Bu, wn + nt * 16 + lm, kk, lane);
        accG[0][nt] = __builtin_amdgcn_wmma_f32_16x16x32_bf16(
            false, a0.v, false, bg.v, (short)0, accG[0][nt], false, false);
        accG[1][nt] = __builtin_amdgcn_wmma_f32_16x16x32_bf16(
            false, a1.v, false, bg.v, (short)0, accG[1][nt], false, false);
        accU[0][nt] = __builtin_amdgcn_wmma_f32_16x16x32_bf16(
            false, a0.v, false, bu.v, (short)0, accU[0][nt], false, false);
        accU[1][nt] = __builtin_amdgcn_wmma_f32_16x16x32_bf16(
            false, a1.v, false, bu.v, (short)0, accU[1][nt], false, false);
      }
    }
    __syncthreads();
  }

  // epilogue: exact GELU on gate path, multiply up path, store bf16
  const int half = lane >> 4, lm = lane & 15;
#pragma unroll
  for (int mt = 0; mt < 2; ++mt)
#pragma unroll
    for (int nt = 0; nt < 4; ++nt) {
      v8f g = accG[mt][nt], u = accU[mt][nt];
#pragma unroll
      for (int r = 0; r < 8; ++r) {
        int m = m0 + wm + mt * 16 + (half << 3) + r;
        int n = n0 + wn + nt * 16 + lm;
        float x   = g[r];
        float act = 0.5f * x * (1.0f + erff(x * 0.70710678118654752f));
        hbuf[(size_t)m * IDIM + n] = (__bf16)(act * u[r]);
      }
    }
}

// ---------------- stage B: out (+)= gate * (h @ W2) ------------------------
__global__ __launch_bounds__(256)
void down_kernel(const __bf16* __restrict__ hbuf,   // [T,I] (L2-resident)
                 const float*  __restrict__ W2,     // [I,H]
                 const float*  __restrict__ gate,   // [T,E]
                 float* __restrict__ out,           // [T,H]
                 int expert, int accumulate) {
  __shared__ alignas(16) __bf16 As[BM * LSTR];
  __shared__ alignas(16) __bf16 Bs[BN * LSTR];

  const int tid  = threadIdx.x;
  const int lane = tid & 31;
  const int wave = tid >> 5;
  const int wm = (wave >> 1) * 32;
  const int wn = (wave & 1) * 64;
  const int m0 = blockIdx.y * BM;
  const int n0 = blockIdx.x * BN;

  const v8f vzero = {0.f,0.f,0.f,0.f,0.f,0.f,0.f,0.f};
  v8f acc[2][4];
#pragma unroll
  for (int i = 0; i < 2; ++i)
#pragma unroll
    for (int j = 0; j < 4; ++j) acc[i][j] = vzero;

  for (int k0 = 0; k0 < IDIM; k0 += BK) {
    // A tile: already bf16 -> pure byte copy via the TDM-era async path.
    // global_load_async_to_lds_b128: VDST = LDS byte offset (low 32 bits of
    // the generic shared pointer), VADDR = 64-bit global address (GV mode).
    // Tracked by ASYNCcnt; drained with s_wait_asynccnt before the barrier.
#pragma unroll
    for (int i = 0; i < 4; ++i) {
      int lin = i * 256 + tid;            // 1024 x 16B slots
      int r = lin >> 3;                   // BK*2/16 = 8 slots per row
      int c = (lin & 7) << 3;             // 8 bf16 = 16B per slot
      const __bf16* g = &hbuf[(size_t)(m0 + r) * IDIM + k0 + c];
      unsigned ldsoff = (unsigned)(size_t)(&As[r * LSTR + c]);
      asm volatile("global_load_async_to_lds_b128 %0, %1, off"
                   :: "v"(ldsoff), "v"(g)
                   : "memory");
    }
    // B tile: f32 -> bf16, transposed [n][k]
#pragma unroll
    for (int i = 0; i < 8; ++i) {
      int lin = i * 256 + tid;
      int k = lin >> 5;
      int c = (lin & 31) << 2;
      f4 a = *(const f4*)&W2[(size_t)(k0 + k) * HDIM + n0 + c];
#pragma unroll
      for (int j = 0; j < 4; ++j)
        Bs[(c + j) * LSTR + k] = (__bf16)a[j];
    }
    if (k0 + BK < IDIM)
      __builtin_prefetch(&W2[(size_t)(k0 + BK + (tid >> 2)) * HDIM + n0], 0, 1);
    asm volatile("s_wait_asynccnt 0x0" ::: "memory");
    __syncthreads();

#pragma unroll
    for (int kk = 0; kk < BK; kk += 32) {
      const int lm = lane & 15;
      FragU a0 = load_frag_a(As, wm + lm,      kk, lane);
      FragU a1 = load_frag_a(As, wm + 16 + lm, kk, lane);
#pragma unroll
      for (int nt = 0; nt < 4; ++nt) {
        FragU b = load_frag_b(Bs, wn + nt * 16 + lm, kk, lane);
        acc[0][nt] = __builtin_amdgcn_wmma_f32_16x16x32_bf16(
            false, a0.v, false, b.v, (short)0, acc[0][nt], false, false);
        acc[1][nt] = __builtin_amdgcn_wmma_f32_16x16x32_bf16(
            false, a1.v, false, b.v, (short)0, acc[1][nt], false, false);
      }
    }
    __syncthreads();
  }

  // epilogue: scale rows by this expert's gate, accumulate into out
  const int half = lane >> 4, lm = lane & 15;
#pragma unroll
  for (int mt = 0; mt < 2; ++mt)
#pragma unroll
    for (int nt = 0; nt < 4; ++nt) {
#pragma unroll
      for (int r = 0; r < 8; ++r) {
        int m = m0 + wm + mt * 16 + (half << 3) + r;
        int n = n0 + wn + nt * 16 + lm;
        float gv = gate[(size_t)m * NEXP + expert];
        float v  = acc[mt][nt][r] * gv;
        float* o = &out[(size_t)m * HDIM + n];
        *o = accumulate ? (*o + v) : v;
      }
    }
}

// ------------------------------ launcher -----------------------------------
extern "C" void kernel_launch(void* const* d_in, const int* in_sizes, int n_in,
                              void* d_out, int out_size, void* d_ws, size_t ws_size,
                              hipStream_t stream) {
  const float* X   = (const float*)d_in[0];  // hidden_states [T,H]
  const float* RI  = (const float*)d_in[1];  // router_input  [T,H]
  const float* SC  = (const float*)d_in[2];  // scale [H]
  const float* PW  = (const float*)d_in[3];  // proj_w [E,H]
  const float* PES = (const float*)d_in[4];  // per_expert_scale [E]
  const float* W1  = (const float*)d_in[5];  // [E,H,I]
  const float* W3  = (const float*)d_in[6];  // [E,H,I]
  const float* W2  = (const float*)d_in[7];  // [E,I,H]
  float* out = (float*)d_out;

  float*  gate = (float*)d_ws;                                   // T*E f32
  __bf16* hbuf = (__bf16*)((char*)d_ws +
                           (size_t)T_TOK * NEXP * sizeof(float)); // T*I bf16

  router_kernel<<<T_TOK / 8, 256, 0, stream>>>(RI, SC, PW, PES, gate);

  dim3 gridA(IDIM / BN, T_TOK / BM);   // (16, 64)
  dim3 gridB(HDIM / BN, T_TOK / BM);   // (16, 64)
  for (int e = 0; e < NEXP; ++e) {
    geglu_up_kernel<<<gridA, 256, 0, stream>>>(
        X, W1 + (size_t)e * HDIM * IDIM, W3 + (size_t)e * HDIM * IDIM, hbuf);
    down_kernel<<<gridB, 256, 0, stream>>>(
        hbuf, W2 + (size_t)e * IDIM * HDIM, gate, out, e, e != 0);
  }
}
